// SelfAttention_47674136985926
// MI455X (gfx1250) — matrix operand order repaired and checked by reading, not compiled, not run on hardware
//
#include <hip/hip_runtime.h>
#include <hip/hip_bf16.h>

typedef __bf16 bf16_t;
typedef __attribute__((ext_vector_type(16))) __bf16 v16bf;
typedef __attribute__((ext_vector_type(8)))  __bf16 v8bf;
typedef __attribute__((ext_vector_type(8)))  float   v8f;

#define NB   2048
#define C_   512
#define CQ_  64
#define NS   64            // spatial N = W*H
#define CN   (C_ * NS)     // 32768 elements per batch item

// LDS row strides (elements), all 16B-multiples in bytes
#define XT_LD 520
#define V_LD  72
#define QK_LD 68
#define T_LD  72

// ---- WMMA fragment helpers (wave32, 16x16x32 bf16) ----------------------
// A (16xK=32) 16-bit layout: lane L holds row M = L%16;
//   lanes<16: K = 0..7 and 16..23 ; lanes>=16: K = 8..15 and 24..31
__device__ __forceinline__ v16bf load_a_frag(const bf16_t* base, int ld,
                                             int m0, int k0, int lane) {
  const int row = m0 + (lane & 15);
  const int kb  = k0 + ((lane >> 4) << 3);
  const bf16_t* p = base + row * ld + kb;
  v8bf lo = *(const v8bf*)(p);        // K = kb .. kb+7
  v8bf hi = *(const v8bf*)(p + 16);   // K = kb+16 .. kb+23
  return __builtin_shufflevector(lo, hi, 0,1,2,3,4,5,6,7,8,9,10,11,12,13,14,15);
}

// B (K=32 x 16): lane L holds col N = L%16; lanes<16 hold K=0..7 and 16..23,
// lanes>=16 hold K=8..15 and 24..31. Source stored transposed S[col][k].
__device__ __forceinline__ v16bf load_b_frag(const bf16_t* baseT, int ld,
                                             int n0, int k0, int lane) {
  const int col = n0 + (lane & 15);
  const int kb  = k0 + ((lane >> 4) << 3);
  const bf16_t* p = baseT + col * ld + kb;
  v8bf lo = *(const v8bf*)(p);
  v8bf hi = *(const v8bf*)(p + 16);
  return __builtin_shufflevector(lo, hi, 0,1,2,3,4,5,6,7,8,9,10,11,12,13,14,15);
}

__device__ __forceinline__ v8f wmma_bf16(v16bf a, v16bf b, v8f c) {
  return __builtin_amdgcn_wmma_f32_16x16x32_bf16(false, a, false, b,
                                                 (short)0, c, false, false);
}

// ---- weight f32 -> bf16 conversion -------------------------------------
__global__ void cvt_f32_bf16_kernel(const float* __restrict__ s,
                                    bf16_t* __restrict__ d, int n) {
  int i = blockIdx.x * 256 + threadIdx.x;
  if (i < n) d[i] = (bf16_t)s[i];
}

// ---- fused attention kernel: one block per batch item ------------------
__global__ __launch_bounds__(256, 1)
void attn_fused_kernel(const float* __restrict__ x,
                       const bf16_t* __restrict__ wqb, const float* __restrict__ bq,
                       const bf16_t* __restrict__ wkb, const float* __restrict__ bk,
                       const bf16_t* __restrict__ wvb, const float* __restrict__ bv,
                       const float* __restrict__ gamma,
                       const float* __restrict__ temperature,
                       const float* __restrict__ lnw, const float* __restrict__ lnb,
                       float* __restrict__ out) {
  __shared__ __align__(32) bf16_t xT[NS * XT_LD];     // x transposed [n][c], bf16
  __shared__ __align__(32) bf16_t vS[C_ * V_LD];      // v [c][m], bf16
  __shared__ __align__(32) float  qf[CQ_ * QK_LD];    // q f32 (later reused for energy)
  __shared__ __align__(32) float  kf[CQ_ * QK_LD];    // k f32
  __shared__ __align__(32) bf16_t qnT[NS * T_LD];     // normalized q, [n][c]
  __shared__ __align__(32) bf16_t knT[NS * T_LD];     // normalized k, [m][c]
  __shared__ __align__(32) bf16_t attnS[NS * T_LD];   // attn [n][m], bf16
  __shared__ float red[2];

  const int b    = blockIdx.x;
  const int tid  = threadIdx.x;
  const int lane = tid & 31;
  const int w    = tid >> 5;
  const float* xb = x + (size_t)b * CN;

  // ---- Phase A: stage x into LDS as transposed bf16 (single HBM read of x) ----
  for (int i = tid; i < CN / 4; i += 256) {
    float4 f = ((const float4*)xb)[i];
    const int c = (4 * i) >> 6;
    const int n = (4 * i) & 63;
    xT[(n + 0) * XT_LD + c] = (bf16_t)f.x;
    xT[(n + 1) * XT_LD + c] = (bf16_t)f.y;
    xT[(n + 2) * XT_LD + c] = (bf16_t)f.z;
    xT[(n + 3) * XT_LD + c] = (bf16_t)f.w;
  }
  if (tid == 0) { red[0] = 0.f; red[1] = 0.f; }
  __syncthreads();

  // ---- Phase B: q = wq*x + bq, k = wk*x + bk (16 tiles each, f32 to LDS) ----
  for (int which = 0; which < 2; ++which) {
    const bf16_t* W    = which ? wkb : wqb;
    const float*  bias = which ? bk  : bq;
    float*        dst  = which ? kf  : qf;
    for (int tt = w; tt < 16; tt += 8) {
      const int m0 = (tt >> 2) << 4;
      const int n0 = (tt & 3) << 4;
      v8f acc = {};
#pragma unroll 4
      for (int kc = 0; kc < C_; kc += 32) {
        v16bf a  = load_a_frag(W, C_, m0, kc, lane);
        v16bf bm = load_b_frag(xT, XT_LD, n0, kc, lane);
        acc = wmma_bf16(a, bm, acc);
      }
      const int col = n0 + (lane & 15);
      const int rb  = m0 + ((lane >> 4) << 3);
#pragma unroll
      for (int r = 0; r < 8; ++r)
        dst[(rb + r) * QK_LD + col] = acc[r] + bias[rb + r];
    }
  }

  // ---- Phase C: v = wv*x + bv (128 tiles, bf16 to LDS) ----
  for (int tt = w * 16; tt < w * 16 + 16; ++tt) {
    const int m0 = (tt >> 2) << 4;
    const int n0 = (tt & 3) << 4;
    v8f acc = {};
#pragma unroll 4
    for (int kc = 0; kc < C_; kc += 32) {
      v16bf a  = load_a_frag(wvb, C_, m0, kc, lane);
      v16bf bm = load_b_frag(xT, XT_LD, n0, kc, lane);
      acc = wmma_bf16(a, bm, acc);
    }
    const int col = n0 + (lane & 15);
    const int rb  = m0 + ((lane >> 4) << 3);
#pragma unroll
    for (int r = 0; r < 8; ++r)
      vS[(rb + r) * V_LD + col] = (bf16_t)(acc[r] + bv[rb + r]);
  }
  __syncthreads();

  // ---- Phase D: L2-normalize q,k rows (over n), write transposed bf16 ----
  if (tid < 128) {
    const int which = tid >> 6;
    const int cr    = tid & 63;
    const float* src = which ? kf : qf;
    bf16_t* dstT     = which ? knT : qnT;
    float ss = 0.f;
#pragma unroll 8
    for (int n = 0; n < NS; ++n) { float t = src[cr * QK_LD + n]; ss += t * t; }
    const float inv = 1.0f / fmaxf(sqrtf(ss), 1e-12f);
#pragma unroll 8
    for (int n = 0; n < NS; ++n)
      dstT[n * T_LD + cr] = (bf16_t)(src[cr * QK_LD + n] * inv);
  }
  __syncthreads();

  // ---- Phase E: energy = (q^T k) / (T + eps), reuse qf as f32 buffer ----
  const float invT = 1.0f / (temperature[0] + 1e-6f);
  float* ef = qf;
  for (int tt = w; tt < 16; tt += 8) {
    const int n0 = (tt >> 2) << 4;   // energy row n
    const int m0 = (tt & 3) << 4;    // energy col m
    v8f acc = {};
#pragma unroll
    for (int kc = 0; kc < CQ_; kc += 32) {
      v16bf a  = load_a_frag(qnT, T_LD, n0, kc, lane);
      v16bf bm = load_b_frag(knT, T_LD, m0, kc, lane);
      acc = wmma_bf16(a, bm, acc);
    }
    const int col = m0 + (lane & 15);
    const int rb  = n0 + ((lane >> 4) << 3);
#pragma unroll
    for (int r = 0; r < 8; ++r)
      ef[(rb + r) * QK_LD + col] = acc[r] * invT;
  }
  __syncthreads();

  // ---- Phase F: softmax rows -> attnS (bf16, [n][m]) ----
  if (tid < 64) {
    const int n = tid;
    float mx = -3.0e38f;
    for (int m = 0; m < NS; ++m) mx = fmaxf(mx, ef[n * QK_LD + m]);
    float s = 0.f;
    for (int m = 0; m < NS; ++m) s += __expf(ef[n * QK_LD + m] - mx);
    const float is = 1.0f / s;
    for (int m = 0; m < NS; ++m)
      attnS[n * T_LD + m] = (bf16_t)(__expf(ef[n * QK_LD + m] - mx) * is);
  }
  __syncthreads();

  // ---- Phase G: out = v * attn^T, 16 tiles/wave kept in registers ----
  v8f oacc[16];
#pragma unroll
  for (int t16 = 0; t16 < 16; ++t16) {
    const int tt = w * 16 + t16;
    const int m0 = (tt >> 2) << 4;
    const int n0 = (tt & 3) << 4;
    v8f acc = {};
#pragma unroll
    for (int kc = 0; kc < NS; kc += 32) {
      v16bf a  = load_a_frag(vS, V_LD, m0, kc, lane);
      v16bf bm = load_b_frag(attnS, T_LD, n0, kc, lane);
      acc = wmma_bf16(a, bm, acc);
    }
    oacc[t16] = acc;
  }

  // ---- residual (x from LDS: no HBM re-read) + LN statistics ----
  const float g    = gamma[0];
  const float og   = 1.0f - g;
  const int   coln = lane & 15;
  const int   hi8  = (lane >> 4) << 3;
  float s1 = 0.f, s2 = 0.f;
#pragma unroll
  for (int t16 = 0; t16 < 16; ++t16) {
    const int tt = w * 16 + t16;
    const int m0 = (tt >> 2) << 4;
    const int nn = ((tt & 3) << 4) + coln;
#pragma unroll
    for (int r = 0; r < 8; ++r) {
      const int cc = m0 + hi8 + r;
      float xv = (float)xT[nn * XT_LD + cc];   // exact when gamma==1 (term vanishes)
      float o2 = g * oacc[t16][r] + og * xv;
      oacc[t16][r] = o2;
      s1 += o2;
      s2 += o2 * o2;
    }
  }
  atomicAdd(&red[0], s1);
  atomicAdd(&red[1], s2);
  __syncthreads();

  const float mean = red[0] * (1.0f / (float)CN);
  const float var  = red[1] * (1.0f / (float)CN) - mean * mean;
  const float rs   = rsqrtf(var + 1e-5f);
  float* ob = out + (size_t)b * CN;
#pragma unroll
  for (int t16 = 0; t16 < 16; ++t16) {
    const int tt = w * 16 + t16;
    const int m0 = (tt >> 2) << 4;
    const int nn = ((tt & 3) << 4) + coln;
#pragma unroll
    for (int r = 0; r < 8; ++r) {
      const int cc  = m0 + hi8 + r;
      const int idx = cc * NS + nn;
      ob[idx] = (oacc[t16][r] - mean) * rs * lnw[idx] + lnb[idx];
    }
  }
}

extern "C" void kernel_launch(void* const* d_in, const int* in_sizes, int n_in,
                              void* d_out, int out_size, void* d_ws, size_t ws_size,
                              hipStream_t stream) {
  const float* x     = (const float*)d_in[0];
  const float* wq    = (const float*)d_in[1];
  const float* bq    = (const float*)d_in[2];
  const float* wk    = (const float*)d_in[3];
  const float* bk    = (const float*)d_in[4];
  const float* wv    = (const float*)d_in[5];
  const float* bv    = (const float*)d_in[6];
  const float* gamma = (const float*)d_in[7];
  const float* temp  = (const float*)d_in[8];
  const float* lnw   = (const float*)d_in[9];
  const float* lnb   = (const float*)d_in[10];
  float* out = (float*)d_out;

  bf16_t* wq_b = (bf16_t*)d_ws;
  bf16_t* wk_b = wq_b + CQ_ * C_;
  bf16_t* wv_b = wk_b + CQ_ * C_;

  cvt_f32_bf16_kernel<<<(CQ_ * C_ + 255) / 256, 256, 0, stream>>>(wq, wq_b, CQ_ * C_);
  cvt_f32_bf16_kernel<<<(CQ_ * C_ + 255) / 256, 256, 0, stream>>>(wk, wk_b, CQ_ * C_);
  cvt_f32_bf16_kernel<<<(C_ * C_ + 255) / 256, 256, 0, stream>>>(wv, wv_b, C_ * C_);

  attn_fused_kernel<<<NB, 256, 0, stream>>>(x, wq_b, bq, wk_b, bk, wv_b, bv,
                                            gamma, temp, lnw, lnb, out);
}
